// MultiheadAttention_48069273977530
// MI455X (gfx1250) — matrix operand order repaired and checked by reading, not compiled, not run on hardware
//
#include <hip/hip_runtime.h>
#include <stdint.h>

// ---------------- problem constants ----------------
#define N_TOK   4096
#define DMODEL  512
#define NHEADS  8
#define HDIM    64
#define SCALE_F 0.125f   // 64^-0.5

// ---------------- WMMA types (CDNA5 / gfx1250, wave32) ----------------
typedef __attribute__((ext_vector_type(16))) __bf16 v16bf;
typedef __attribute__((ext_vector_type(8)))  __bf16 v8bf;
typedef __attribute__((ext_vector_type(8)))  float  v8f;

static __device__ __forceinline__ v8f wmma_bf16(v16bf a, v16bf b, v8f c) {
  // D(16x16 f32) = A(16x32 bf16) x B(32x16 bf16) + C
  return __builtin_amdgcn_wmma_f32_16x16x32_bf16(
      /*neg_a=*/false, a, /*neg_b=*/false, b,
      /*c_mod=*/(short)0, c, /*reuse_a=*/false, /*reuse_b=*/false);
}

// ---------- fragment loaders (contiguous k order, same in A and B) ----------
// A (16x32): lane row = lane&15; elems 0..15 -> K = kb..kb+15 (contiguous)
// with kb = (lane>>4)*16.
static __device__ __forceinline__ v16bf load_a_f32(const float* __restrict__ base,
                                                   int ld, int row, int k0, int lane) {
  int kb = (lane >> 4) * 16;
  const float* p = base + (size_t)row * ld + k0 + kb;
  float t[16];
  *(float4*)(t + 0)  = *(const float4*)(p + 0);
  *(float4*)(t + 4)  = *(const float4*)(p + 4);
  *(float4*)(t + 8)  = *(const float4*)(p + 8);
  *(float4*)(t + 12) = *(const float4*)(p + 12);
  v16bf a;
#pragma unroll
  for (int i = 0; i < 16; ++i) a[i] = (__bf16)t[i];
  return a;
}

static __device__ __forceinline__ v16bf load_a_bf16(const __bf16* __restrict__ base,
                                                    int ld, int row, int k0, int lane) {
  int kb = (lane >> 4) * 16;
  const __bf16* p = base + (size_t)row * ld + k0 + kb;
  v8bf lo = *(const v8bf*)(p);
  v8bf hi = *(const v8bf*)(p + 8);
  v16bf a;
#pragma unroll
  for (int i = 0; i < 8; ++i) { a[i] = lo[i]; a[8 + i] = hi[i]; }
  return a;
}

// B (32x16) built from row-major W so that B[k][j] = W[wrow=j][k0+koff+i]
// lane: col = lane&15, koff = (lane>>4)*16, elems i = K koff..koff+15 (contiguous).
static __device__ __forceinline__ v16bf load_b_f32(const float* __restrict__ W,
                                                   int ld, int wrow, int k0, int lane) {
  int koff = (lane >> 4) * 16;
  const float* p = W + (size_t)wrow * ld + k0 + koff;
  float t[16];
  *(float4*)(t + 0)  = *(const float4*)(p + 0);
  *(float4*)(t + 4)  = *(const float4*)(p + 4);
  *(float4*)(t + 8)  = *(const float4*)(p + 8);
  *(float4*)(t + 12) = *(const float4*)(p + 12);
  v16bf b;
#pragma unroll
  for (int i = 0; i < 16; ++i) b[i] = (__bf16)t[i];
  return b;
}

// half-wave (16 lane) row reductions — rows of a C tile live across one 16-lane half
static __device__ __forceinline__ float rowmax16(float x) {
  x = fmaxf(x, __shfl_xor(x, 1, 16));
  x = fmaxf(x, __shfl_xor(x, 2, 16));
  x = fmaxf(x, __shfl_xor(x, 4, 16));
  x = fmaxf(x, __shfl_xor(x, 8, 16));
  return x;
}
static __device__ __forceinline__ float rowsum16(float x) {
  x += __shfl_xor(x, 1, 16);
  x += __shfl_xor(x, 2, 16);
  x += __shfl_xor(x, 4, 16);
  x += __shfl_xor(x, 8, 16);
  return x;
}

// ================= kernel 1: fused Q/K/V projections (f32 -> bf16) =================
// Each wave computes a 16x64 tile of X @ W^T + b (one A fragment reused by 4 B
// fragments -> 4 WMMAs per A load).  V is stored transposed (Vt[d][n]) so the
// attention kernel's P*V B-fragments are contiguous loads.
__global__ __launch_bounds__(256) void qkv_proj_kernel(
    const float* __restrict__ q, const float* __restrict__ k, const float* __restrict__ v,
    const float* __restrict__ wq, const float* __restrict__ bq,
    const float* __restrict__ wk, const float* __restrict__ bk,
    const float* __restrict__ wv, const float* __restrict__ bv,
    __bf16* __restrict__ Qp, __bf16* __restrict__ Kp, __bf16* __restrict__ Vt) {
  int wid  = threadIdx.x >> 5;
  int lane = threadIdx.x & 31;
  int tile = blockIdx.x * 8 + wid;           // 3 * 2048 wave-tiles total
  int which = tile / 2048;                   // 0=Q 1=K 2=V
  int t = tile - which * 2048;
  int n0 = (t >> 3) * 16;                    // 256 row tiles
  int o0 = (t & 7) * 64;                     // 8 column groups of 64

  const float* X; const float* W; const float* B;
  if      (which == 0) { X = q; W = wq; B = bq; }
  else if (which == 1) { X = k; W = wk; B = bk; }
  else                 { X = v; W = wv; B = bv; }

  int col = lane & 15;
  v8f acc[4] = {};
#pragma unroll 2
  for (int k0 = 0; k0 < DMODEL; k0 += 32) {
    v16bf a = load_a_f32(X, DMODEL, n0 + col, k0, lane);
#pragma unroll
    for (int tc = 0; tc < 4; ++tc) {
      v16bf b = load_b_f32(W, DMODEL, o0 + tc * 16 + col, k0, lane);
      acc[tc] = wmma_bf16(a, b, acc[tc]);
    }
  }
  int rbase = n0 + (lane >> 4) * 8;
#pragma unroll
  for (int tc = 0; tc < 4; ++tc) {
    int oc = o0 + tc * 16 + col;
    float bias = B[oc];
    if (which == 2) {
#pragma unroll
      for (int r = 0; r < 8; ++r)
        Vt[(size_t)oc * N_TOK + (rbase + r)] = (__bf16)(acc[tc][r] + bias);
    } else {
      __bf16* out = (which == 0) ? Qp : Kp;
#pragma unroll
      for (int r = 0; r < 8; ++r)
        out[(size_t)(rbase + r) * DMODEL + oc] = (__bf16)(acc[tc][r] + bias);
    }
  }
}

// ================= kernel 2: flash attention per head =================
// One block = one head, 8 waves = 8 adjacent 16-row query blocks.  Per 32-key
// iteration the block cooperatively stages K(32x64) and V(64x32) into LDS once,
// then every wave runs: S via 4 WMMAs (from LDS), masked online softmax (shfl
// reductions), P transposed through per-wave LDS, P*V via 4 WMMAs (from LDS).
__global__ __launch_bounds__(256) void attention_kernel(
    const __bf16* __restrict__ Qp, const __bf16* __restrict__ Kp,
    const __bf16* __restrict__ Vt, const uint8_t* __restrict__ mask,
    __bf16* __restrict__ O) {
  __shared__ __bf16 kt[32 * 64];             // K tile  [key][d]   4 KB
  __shared__ __bf16 vt[64 * 32];             // V tile  [d][key]   4 KB
  __shared__ __bf16 pbuf[8][16 * 32];        // per-wave P staging 8 KB

  int tid  = threadIdx.x;
  int wid  = tid >> 5;
  int lane = tid & 31;
  int h    = blockIdx.x >> 5;                // 8 heads
  int rb   = (blockIdx.x & 31) * 8 + wid;    // 256 query row blocks per head
  int n0   = rb * 16;
  int col  = lane & 15;
  int lh   = lane >> 4;                      // lane half
  int rbase = n0 + lh * 8;
  int koff = lh * 16;

  // Q A-fragments for head h (K = 0..31 and 32..63), loaded once
  v16bf aq0 = load_a_bf16(Qp, DMODEL, n0 + col, h * HDIM + 0,  lane);
  v16bf aq1 = load_a_bf16(Qp, DMODEL, n0 + col, h * HDIM + 32, lane);

  float m[8], l[8];
  v8f oa0 = {}, oa1 = {}, oa2 = {}, oa3 = {};
#pragma unroll
  for (int r = 0; r < 8; ++r) { m[r] = -3.0e38f; l[r] = 0.0f; }

  __bf16* pb = &pbuf[wid][0];

  // staging addresses: each of 256 threads moves one 16B chunk for K and V
  int kflat = tid * 8;                       // 2048 bf16 = 32x64
  int krow  = kflat >> 6, kcol = kflat & 63;
  int vflat = tid * 8;                       // 2048 bf16 = 64x32
  int vrow  = vflat >> 5, vcol = vflat & 31;

#pragma unroll 1
  for (int m0 = 0; m0 < N_TOK; m0 += 32) {
    __syncthreads();                         // previous tiles fully consumed
    *(uint4*)(kt + krow * 64 + kcol) =
        *(const uint4*)(Kp + (size_t)(m0 + krow) * DMODEL + h * HDIM + kcol);
    *(uint4*)(vt + vrow * 32 + vcol) =
        *(const uint4*)(Vt + ((size_t)h * HDIM + vrow) * N_TOK + m0 + vcol);
    __syncthreads();

    // ---- S = Q K^T for the 32-key block (two 16-key column tiles) ----
    v8f s0 = {}, s1 = {};
    {
      const __bf16* kr0 = kt + col * 64 + koff;
      s0 = wmma_bf16(aq0, *(const v16bf*)(kr0 + 0),  s0);
      s0 = wmma_bf16(aq1, *(const v16bf*)(kr0 + 32), s0);
      const __bf16* kr1 = kt + (16 + col) * 64 + koff;
      s1 = wmma_bf16(aq0, *(const v16bf*)(kr1 + 0),  s1);
      s1 = wmma_bf16(aq1, *(const v16bf*)(kr1 + 32), s1);
    }
    // ---- masked online softmax update ----
#pragma unroll
    for (int r = 0; r < 8; ++r) {
      int row = rbase + r;
      const uint8_t* mrow = mask + (size_t)row * N_TOK + m0;
      float v0 = s0[r] * SCALE_F;
      float v1 = s1[r] * SCALE_F;
      if (mrow[col])      v0 = -1.0e6f;
      if (mrow[16 + col]) v1 = -1.0e6f;
      float mx   = rowmax16(fmaxf(v0, v1));
      float mnew = fmaxf(m[r], mx);
      float alpha = __expf(m[r] - mnew);
      float p0 = __expf(v0 - mnew);
      float p1 = __expf(v1 - mnew);
      l[r] = l[r] * alpha + rowsum16(p0 + p1);
      m[r] = mnew;
      oa0[r] *= alpha; oa1[r] *= alpha; oa2[r] *= alpha; oa3[r] *= alpha;
      int rl = lh * 8 + r;                   // C-layout row -> LDS row-major P
      pb[rl * 32 + col]      = (__bf16)p0;
      pb[rl * 32 + 16 + col] = (__bf16)p1;
    }
    __builtin_amdgcn_wave_barrier();         // keep DS store->load order across lanes
    // ---- reload P as an A-fragment (transpose through LDS) ----
    v16bf ap;
    {
      int prow = col;
      int kb = lh * 16;
#pragma unroll
      for (int i = 0; i < 8; ++i) {
        ap[i]     = pb[prow * 32 + kb + i];
        ap[8 + i] = pb[prow * 32 + kb + 8 + i];
      }
    }
    __builtin_amdgcn_wave_barrier();
    // ---- O += P(16x32) @ V(32x64) from the LDS V tile [d][key] ----
    const __bf16* vb = vt + col * 32 + koff;
    oa0 = wmma_bf16(ap, *(const v16bf*)(vb + 0  * 16 * 32), oa0);
    oa1 = wmma_bf16(ap, *(const v16bf*)(vb + 1  * 16 * 32), oa1);
    oa2 = wmma_bf16(ap, *(const v16bf*)(vb + 2  * 16 * 32), oa2);
    oa3 = wmma_bf16(ap, *(const v16bf*)(vb + 3  * 16 * 32), oa3);
  }

  // ---- finalize: O / l, store bf16 (N x 512 row-major) ----
#pragma unroll
  for (int r = 0; r < 8; ++r) {
    float inv = 1.0f / l[r];
    __bf16* op = O + (size_t)(rbase + r) * DMODEL + h * HDIM + col;
    op[0]  = (__bf16)(oa0[r] * inv);
    op[16] = (__bf16)(oa1[r] * inv);
    op[32] = (__bf16)(oa2[r] * inv);
    op[48] = (__bf16)(oa3[r] * inv);
  }
}

// ================= kernel 3: output projection (bf16 -> f32) =================
// 16x64 output per wave: one A fragment reused by 4 B fragments.
__global__ __launch_bounds__(256) void out_proj_kernel(
    const __bf16* __restrict__ O, const float* __restrict__ wo,
    const float* __restrict__ bo, float* __restrict__ out) {
  int wid  = threadIdx.x >> 5;
  int lane = threadIdx.x & 31;
  int tile = blockIdx.x * 8 + wid;           // 2048 wave-tiles
  int n0 = (tile >> 3) * 16;
  int o0 = (tile & 7) * 64;
  int col = lane & 15;

  v8f acc[4] = {};
#pragma unroll 2
  for (int k0 = 0; k0 < DMODEL; k0 += 32) {
    v16bf a = load_a_bf16(O, DMODEL, n0 + col, k0, lane);
#pragma unroll
    for (int tc = 0; tc < 4; ++tc) {
      v16bf b = load_b_f32(wo, DMODEL, o0 + tc * 16 + col, k0, lane);
      acc[tc] = wmma_bf16(a, b, acc[tc]);
    }
  }
  int rbase = n0 + (lane >> 4) * 8;
#pragma unroll
  for (int tc = 0; tc < 4; ++tc) {
    int oc = o0 + tc * 16 + col;
    float bias = bo[oc];
#pragma unroll
    for (int r = 0; r < 8; ++r)
      out[(size_t)(rbase + r) * DMODEL + oc] = acc[tc][r] + bias;
  }
}

// ================= host launcher =================
extern "C" void kernel_launch(void* const* d_in, const int* in_sizes, int n_in,
                              void* d_out, int out_size, void* d_ws, size_t ws_size,
                              hipStream_t stream) {
  const float*   q    = (const float*)d_in[0];
  const float*   k    = (const float*)d_in[1];
  const float*   v    = (const float*)d_in[2];
  const uint8_t* mask = (const uint8_t*)d_in[3];   // jax bool -> 1 byte
  const float* wq_w = (const float*)d_in[4];
  const float* wq_b = (const float*)d_in[5];
  const float* wk_w = (const float*)d_in[6];
  const float* wk_b = (const float*)d_in[7];
  const float* wv_w = (const float*)d_in[8];
  const float* wv_b = (const float*)d_in[9];
  const float* wo_w = (const float*)d_in[10];
  const float* wo_b = (const float*)d_in[11];
  (void)in_sizes; (void)n_in; (void)out_size; (void)ws_size;

  // workspace: Qp(4MB) | Kp(4MB) | Vt(4MB, transposed [d][n]) | O(4MB)  — all bf16
  char* ws = (char*)d_ws;
  __bf16* Qp = (__bf16*)(ws + (size_t)0);
  __bf16* Kp = (__bf16*)(ws + (size_t)4  * 1024 * 1024);
  __bf16* Vt = (__bf16*)(ws + (size_t)8  * 1024 * 1024);
  __bf16* Oa = (__bf16*)(ws + (size_t)12 * 1024 * 1024);

  qkv_proj_kernel<<<768, 256, 0, stream>>>(q, k, v, wq_w, wq_b, wk_w, wk_b,
                                           wv_w, wv_b, Qp, Kp, Vt);
  attention_kernel<<<256, 256, 0, stream>>>(Qp, Kp, Vt, mask, Oa);
  out_proj_kernel<<<256, 256, 0, stream>>>(Oa, wo_w, wo_b, (float*)d_out);
}
